// GraphTab_v2_73899207295335
// MI455X (gfx1250) — compile-verified, matrix operand order checked by reading, not run on hardware
//
#include <hip/hip_runtime.h>
#include <hip/hip_bf16.h>
#include <math.h>
#include <stdint.h>

typedef __attribute__((ext_vector_type(16))) _Float16     v16h;
typedef __attribute__((ext_vector_type(8)))  float        v8f;
typedef __attribute__((ext_vector_type(2)))  float        v2f;
typedef __attribute__((ext_vector_type(4)))  unsigned int v4u;
typedef __attribute__((ext_vector_type(8)))  int          v8i;
typedef __attribute__((ext_vector_type(4)))  int          v4i;

#define N_NODES   64000
#define N_GRAPHS  64
#define NODES_PER 1000

#if __has_builtin(__builtin_amdgcn_tensor_load_to_lds)
#define USE_TDM 1
#if __has_include(<hip/amd_detail/amd_gfx1250_TDM.h>)
#define TDM_ARGS6 1
#endif
#endif

// ---------------------------------------------------------------------------
// utility
// ---------------------------------------------------------------------------
__global__ void fill_kernel(float* __restrict__ p, int n, float v) {
    int i = blockIdx.x * blockDim.x + threadIdx.x;
    if (i < n) p[i] = v;
}

__device__ __forceinline__ void atomicMaxFloat(float* addr, float val) {
    // valid with buffer initialized to -inf
    if (val >= 0.0f) atomicMax((int*)addr, __float_as_int(val));
    else             atomicMin((unsigned int*)addr, __float_as_uint(val));
}

#if USE_TDM
// ---------------------------------------------------------------------------
// Tensor Data Mover: 2D tile load Global->LDS, f32 elements.
// D# packing per CDNA5 ISA ch.8: group0 = {count|flags, lds_addr, gaddr, type},
// group1 = {data_size, tensor_dim0/1, tile_dim0/1, tensor_dim0_stride}.
// ---------------------------------------------------------------------------
__device__ __forceinline__ void tdm_load_2d(void* lds, const void* gptr,
                                            unsigned td0, unsigned td1,
                                            unsigned tile0, unsigned tile1,
                                            unsigned long long stride0_elems) {
    unsigned lds_addr = (unsigned)(uintptr_t)lds;            // low 32 = LDS offset
    unsigned long long ga = (unsigned long long)(uintptr_t)gptr;
    v4u g0;
    g0[0] = 1u;                                              // count=1, user mode
    g0[1] = lds_addr;
    g0[2] = (unsigned)(ga & 0xffffffffu);
    g0[3] = (unsigned)((ga >> 32) & 0x01ffffffu) | (2u << 30);   // type=2 (image)
    v8i g1;
    g1[0] = (int)(2u << 16);                                 // data_size=4B, no multicast
    g1[1] = (int)((td0 & 0xffffu) << 16);                    // tensor_dim0[15:0]
    g1[2] = (int)(((td0 >> 16) & 0xffffu) | ((td1 & 0xffffu) << 16));
    g1[3] = (int)(((td1 >> 16) & 0xffffu) | ((tile0 & 0xffffu) << 16));
    g1[4] = (int)(tile1 & 0xffffu);                          // tile_dim1; tile_dim2=0
    g1[5] = (int)(stride0_elems & 0xffffffffu);              // dim0 stride (elems)
    g1[6] = (int)((stride0_elems >> 32) & 0xffffu);          // dim1 stride = 0
    g1[7] = 0;
    v4i z4 = {0, 0, 0, 0};
#if TDM_ARGS6
    v8i z8 = {0, 0, 0, 0, 0, 0, 0, 0};
    __builtin_amdgcn_tensor_load_to_lds(g0, g1, z4, z4, z8, 0);
#else
    __builtin_amdgcn_tensor_load_to_lds(g0, g1, z4, z4, 0);
#endif
}
#endif

// ---------------------------------------------------------------------------
// Layer-1 GEMM: h1[64000,256] = X[64000,4] @ W1[4,256]   (K=4 -> f32 WMMA)
// grid (M/16, 2), block 256 (8 waves); wave handles one 16x16 N tile
// ---------------------------------------------------------------------------
#if __has_builtin(__builtin_amdgcn_wmma_f32_16x16x4_f32)
__global__ __launch_bounds__(256) void gemm1_kernel(const float* __restrict__ X,
                                                    const float* __restrict__ W,
                                                    float* __restrict__ Y, int M) {
    int tid  = threadIdx.x;
    int wave = tid >> 5, lane = tid & 31;
    int hi   = lane >> 4;
    int row  = lane & 15;
    int m0   = blockIdx.x * 16;
    int nt   = blockIdx.y * 8 + wave;
    int n    = nt * 16 + row;
    v2f a, b;
    a.x = X[(size_t)(m0 + row) * 4 + hi * 2 + 0];
    a.y = X[(size_t)(m0 + row) * 4 + hi * 2 + 1];
    b.x = W[(size_t)(hi * 2 + 0) * 256 + n];
    b.y = W[(size_t)(hi * 2 + 1) * 256 + n];
    v8f acc = {};
    acc = __builtin_amdgcn_wmma_f32_16x16x4_f32(false, a, false, b,
                                                (short)0, acc, false, false);
#pragma unroll
    for (int r = 0; r < 8; ++r)
        Y[(size_t)(m0 + r + hi * 8) * 256 + n] = acc[r];
}
#else
__global__ __launch_bounds__(256) void gemm1_kernel(const float* __restrict__ X,
                                                    const float* __restrict__ W,
                                                    float* __restrict__ Y, int M) {
    int m0 = blockIdx.x * 16, c0 = blockIdx.y * 128;
#pragma unroll
    for (int i = 0; i < 8; ++i) {
        int lin = threadIdx.x * 8 + i;
        int r = m0 + (lin >> 7), c = c0 + (lin & 127);
        float acc = 0.f;
#pragma unroll
        for (int k = 0; k < 4; ++k) acc += X[(size_t)r * 4 + k] * W[k * 256 + c];
        Y[(size_t)r * 256 + c] = acc;
    }
}
#endif

// ---------------------------------------------------------------------------
// Layer-2 GEMM: Y[M,128] = X[M,256] @ W[256,128].
// Block = 128 rows x 128 cols, 256 threads (8 waves). Wave w owns rows
// [w*16, w*16+16) and all 8 N tiles -> 8 f32 accumulators, 8 WMMAs / K-step.
// K tiles (f32) staged to LDS by the Tensor Data Mover, double-buffered,
// overlap via TENSORcnt; f32->f16 conversion during fragment build.
// ---------------------------------------------------------------------------
__global__ __launch_bounds__(256) void gemm2_wmma_kernel(const float* __restrict__ X,
                                                         const float* __restrict__ W,
                                                         float* __restrict__ Y, int M) {
    __shared__ float Asb[2][128 * 32];   // 16 KB each: 128 rows x 32 k
    __shared__ float Bsb[2][32 * 128];   // 16 KB each: 32 k x 128 n
    const int tid  = threadIdx.x;
    const int wave = tid >> 5, lane = tid & 31;
    const int hi   = lane >> 4;
    const int row  = lane & 15;
    const int m0   = blockIdx.x * 128;
    v8f acc[8] = {};

#if USE_TDM
    if (wave == 0) {
        tdm_load_2d(&Asb[0][0], X + (size_t)m0 * 256, 32, 128, 32, 128, 256);
        tdm_load_2d(&Bsb[0][0], W, 128, 32, 128, 32, 128);
    }
#endif
    for (int t = 0; t < 8; ++t) {
        const int ko = t * 32;
#if USE_TDM
        if (wave == 0) {
            if (t + 1 < 8) {
                // prefetch next K tile into other buffer, then wait for current
                tdm_load_2d(&Asb[(t + 1) & 1][0],
                            X + (size_t)m0 * 256 + (ko + 32), 32, 128, 32, 128, 256);
                tdm_load_2d(&Bsb[(t + 1) & 1][0],
                            W + (size_t)(ko + 32) * 128, 128, 32, 128, 32, 128);
                __builtin_amdgcn_s_wait_tensorcnt(2);
            } else {
                __builtin_amdgcn_s_wait_tensorcnt(0);
            }
        }
        __syncthreads();
#else
        // manual cooperative staging fallback
        for (int i = tid; i < 128 * 32; i += 256) {
            int r = i >> 5, k = i & 31;
            Asb[t & 1][i] = X[(size_t)(m0 + r) * 256 + ko + k];
        }
        for (int i = tid; i < 32 * 128; i += 256) {
            int k = i >> 7, n = i & 127;
            Bsb[t & 1][i] = W[(size_t)(ko + k) * 128 + n];
        }
        __syncthreads();
#endif
        const float* A32 = &Asb[t & 1][0];
        const float* B32 = &Bsb[t & 1][0];
        const int arow = wave * 16 + row;
        v16h a;
#pragma unroll
        for (int j = 0; j < 16; ++j) {
            // per-lane K index for 16-bit 16x32 fragments (ISA 7.12.2)
            int k = ((j >> 3) << 4) + (hi << 3) + (j & 7);
            a[j] = (_Float16)A32[arow * 32 + k];
        }
#pragma unroll
        for (int nt = 0; nt < 8; ++nt) {
            v16h b;
#pragma unroll
            for (int j = 0; j < 16; ++j) {
                int k = ((j >> 3) << 4) + (hi << 3) + (j & 7);
                b[j] = (_Float16)B32[k * 128 + nt * 16 + row];
            }
            acc[nt] = __builtin_amdgcn_wmma_f32_16x16x32_f16(false, a, false, b,
                                                             (short)0, acc[nt],
                                                             false, false);
        }
        __syncthreads();
    }
#pragma unroll
    for (int nt = 0; nt < 8; ++nt) {
#pragma unroll
        for (int r = 0; r < 8; ++r)
            Y[(size_t)(m0 + wave * 16 + r + hi * 8) * 128 + nt * 16 + row] = acc[nt][r];
    }
}

// ---------------------------------------------------------------------------
// per-node attention scores: s[n] = h[n].a_src, d[n] = h[n].a_dst
// ---------------------------------------------------------------------------
__global__ __launch_bounds__(256) void scores_kernel(const float* __restrict__ H,
                                                     const float* __restrict__ av,
                                                     const float* __restrict__ bv,
                                                     float* __restrict__ s,
                                                     float* __restrict__ d,
                                                     int N, int C) {
    int wave = threadIdx.x >> 5, lane = threadIdx.x & 31;
    int node = blockIdx.x * 8 + wave;
    if (node >= N) return;
    float as = 0.f, ad = 0.f;
    for (int c = lane; c < C; c += 32) {
        float h = H[(size_t)node * C + c];
        as += h * av[c];
        ad += h * bv[c];
    }
#pragma unroll
    for (int off = 16; off > 0; off >>= 1) {
        as += __shfl_xor(as, off, 32);
        ad += __shfl_xor(ad, off, 32);
    }
    if (lane == 0) { s[node] = as; d[node] = ad; }
}

// ---------------------------------------------------------------------------
// edge pass 1: e = leaky(s[src]+d[dst]); segment max into m[dst]
// ---------------------------------------------------------------------------
__global__ void edge_max_kernel(const long long* __restrict__ esrc,
                                const long long* __restrict__ edst,
                                const float* __restrict__ s, const float* __restrict__ d,
                                float* __restrict__ ew, float* __restrict__ m,
                                int E0, int E) {
    int e = blockIdx.x * blockDim.x + threadIdx.x;
    if (e >= E) return;
    int src, dst;
    if (e < E0) { src = (int)esrc[e]; dst = (int)edst[e]; }
    else        { src = dst = e - E0; }                 // self loops appended
    float v = s[src] + d[dst];
    v = (v >= 0.f) ? v : 0.2f * v;                      // LeakyReLU(0.2)
    ew[e] = v;
    atomicMaxFloat(&m[dst], v);
}

// ---------------------------------------------------------------------------
// edge pass 2: p = exp(e - m[dst]); z[dst] += p
// ---------------------------------------------------------------------------
__global__ void edge_exp_kernel(const long long* __restrict__ esrc,
                                const long long* __restrict__ edst,
                                float* __restrict__ ew, const float* __restrict__ m,
                                float* __restrict__ z, int E0, int E) {
    int e = blockIdx.x * blockDim.x + threadIdx.x;
    if (e >= E) return;
    int dst = (e < E0) ? (int)edst[e] : (e - E0);
    float p = expf(ew[e] - m[dst]);
    ew[e] = p;
    atomicAdd(&z[dst], p);
}

// ---------------------------------------------------------------------------
// edge pass 3: out[dst] += (p/z[dst]) * h[src]   (wave per edge)
// ---------------------------------------------------------------------------
__global__ __launch_bounds__(256) void aggregate_kernel(const long long* __restrict__ esrc,
                                                        const long long* __restrict__ edst,
                                                        const float* __restrict__ ew,
                                                        const float* __restrict__ z,
                                                        const float* __restrict__ H,
                                                        float* __restrict__ out,
                                                        int E0, int E, int C) {
    int wave = threadIdx.x >> 5, lane = threadIdx.x & 31;
    int e = blockIdx.x * 8 + wave;
    if (e >= E) return;
    int src, dst;
    if (e < E0) { src = (int)esrc[e]; dst = (int)edst[e]; }
    else        { src = dst = e - E0; }
    float alpha = ew[e] / z[dst];
    const float* hrow = H + (size_t)src * C;
    float* orow = out + (size_t)dst * C;
    __builtin_prefetch(hrow, 0, 3);
    for (int c = lane; c < C; c += 32)
        atomicAdd(&orow[c], alpha * hrow[c]);
}

// ---------------------------------------------------------------------------
// x = relu(x + bias[c])
// ---------------------------------------------------------------------------
__global__ void bias_relu_kernel(float* __restrict__ X, const float* __restrict__ b,
                                 int N, int C) {
    int i = blockIdx.x * blockDim.x + threadIdx.x;
    if (i >= N * C) return;
    float v = X[i] + b[i % C];
    X[i] = fmaxf(v, 0.f);
}

// ---------------------------------------------------------------------------
// global max pool: [64000,128] -> [64,128]   (grid 64, block 128)
// ---------------------------------------------------------------------------
__global__ void pool_kernel(const float* __restrict__ X, float* __restrict__ Y) {
    int g = blockIdx.x, c = threadIdx.x;
    float mx = -3.402823466e38f;
    const float* base = X + (size_t)g * NODES_PER * 128 + c;
    for (int i = 0; i < NODES_PER; ++i)
        mx = fmaxf(mx, base[(size_t)i * 128]);
    Y[g * 128 + c] = mx;
}

// ---------------------------------------------------------------------------
// small head GEMM: Y[64,M] = A[64,K] @ W[K,M] + bias ; act: 0=none 1=relu
// ---------------------------------------------------------------------------
__global__ void head_gemm_kernel(const float* __restrict__ A, const float* __restrict__ W,
                                 const float* __restrict__ bias, float* __restrict__ Y,
                                 int K, int M, int act) {
    int i = blockIdx.x * blockDim.x + threadIdx.x;
    if (i >= 64 * M) return;
    int r = i / M, c = i % M;
    float acc = bias[c];
    for (int k = 0; k < K; ++k) acc += A[r * K + k] * W[k * M + c];
    if (act == 1) acc = fmaxf(acc, 0.f);
    Y[i] = acc;
}

// ---------------------------------------------------------------------------
// BatchNorm over 64 rows (biased var) + act: 0=none 1=relu 2=elu ; in place
// ---------------------------------------------------------------------------
__global__ void bn_kernel(float* __restrict__ X, const float* __restrict__ g,
                          const float* __restrict__ b, int M, int act) {
    int c = blockIdx.x * blockDim.x + threadIdx.x;
    if (c >= M) return;
    float mu = 0.f;
    for (int r = 0; r < 64; ++r) mu += X[r * M + c];
    mu *= (1.f / 64.f);
    float var = 0.f;
    for (int r = 0; r < 64; ++r) { float t = X[r * M + c] - mu; var += t * t; }
    var *= (1.f / 64.f);
    float inv = rsqrtf(var + 1e-5f) * g[c];
    for (int r = 0; r < 64; ++r) {
        float v = (X[r * M + c] - mu) * inv + b[c];
        if (act == 1)      v = fmaxf(v, 0.f);
        else if (act == 2) v = (v > 0.f) ? v : (expf(v) - 1.f);
        X[r * M + c] = v;
    }
}

__global__ void concat_kernel(const float* __restrict__ A, const float* __restrict__ B,
                              float* __restrict__ C) {
    int i = blockIdx.x * blockDim.x + threadIdx.x;
    if (i >= 64 * 256) return;
    int r = i >> 8, c = i & 255;
    C[i] = (c < 128) ? A[r * 128 + c] : B[r * 128 + (c - 128)];
}

__global__ void final_kernel(const float* __restrict__ F, const float* __restrict__ w,
                             const float* __restrict__ b, float* __restrict__ y) {
    int r = threadIdx.x;
    if (r >= 64) return;
    float acc = b[0];
    for (int c = 0; c < 64; ++c) acc += F[r * 64 + c] * w[c];
    y[r] = acc;
}

// ---------------------------------------------------------------------------
// launcher
// ---------------------------------------------------------------------------
extern "C" void kernel_launch(void* const* d_in, const int* in_sizes, int n_in,
                              void* d_out, int out_size, void* d_ws, size_t ws_size,
                              hipStream_t stream) {
    const float*     cell_x = (const float*)d_in[0];
    const long long* eidx   = (const long long*)d_in[1];
    const float*     drug   = (const float*)d_in[3];
    const float *W1 = (const float*)d_in[4],  *as1 = (const float*)d_in[5];
    const float *ad1 = (const float*)d_in[6], *b1  = (const float*)d_in[7];
    const float *W2 = (const float*)d_in[8],  *as2 = (const float*)d_in[9];
    const float *ad2 = (const float*)d_in[10],*b2  = (const float*)d_in[11];
    const float *cl1_w=(const float*)d_in[12],*cl1_b=(const float*)d_in[13];
    const float *cbn1_g=(const float*)d_in[14],*cbn1_b=(const float*)d_in[15];
    const float *cl2_w=(const float*)d_in[16],*cl2_b=(const float*)d_in[17];
    const float *d1_w=(const float*)d_in[18], *d1_b=(const float*)d_in[19];
    const float *dbn1_g=(const float*)d_in[20],*dbn1_b=(const float*)d_in[21];
    const float *d2_w=(const float*)d_in[22], *d2_b=(const float*)d_in[23];
    const float *dbn2_g=(const float*)d_in[24],*dbn2_b=(const float*)d_in[25];
    const float *f1_w=(const float*)d_in[26], *f1_b=(const float*)d_in[27];
    const float *fbn1_g=(const float*)d_in[28],*fbn1_b=(const float*)d_in[29];
    const float *f2_w=(const float*)d_in[30], *f2_b=(const float*)d_in[31];
    const float *fbn2_g=(const float*)d_in[32],*fbn2_b=(const float*)d_in[33];
    const float *f3_w=(const float*)d_in[34], *f3_b=(const float*)d_in[35];

    const int N  = in_sizes[0] / 4;      // 64000
    const int E0 = in_sizes[1] / 2;      // 1,024,000
    const int E  = E0 + N;               // + self loops
    const long long* esrc = eidx;
    const long long* edst = eidx + E0;

    // workspace layout (floats)
    float* ws = (float*)d_ws;
    size_t off = 0;
    float* h1   = ws + off; off += (size_t)N * 256;   // reused: h2 / o2 overlay
    float* o1   = ws + off; off += (size_t)N * 256;   // x1 after bias+relu
    float* sbuf = ws + off; off += N;
    float* dbuf = ws + off; off += N;
    float* mbuf = ws + off; off += N;
    float* zbuf = ws + off; off += N;
    float* ew   = ws + off; off += E;                 // edge e / p values
    float* x3   = ws + off; off += 64 * 128;
    float* t1   = ws + off; off += 64 * 128;
    float* cemb = ws + off; off += 64 * 128;
    float* u1   = ws + off; off += 64 * 128;
    float* demb = ws + off; off += 64 * 128;
    float* cat  = ws + off; off += 64 * 256;
    float* fm1  = ws + off; off += 64 * 128;
    float* fm2  = ws + off; off += 64 * 64;
    float* h2 = h1;                     // [N,128]
    float* o2 = h1 + (size_t)N * 128;   // [N,128]

    const float NEG_INF = -3.402823466e38f;
    dim3 blk(256);
    int egrid = (E + 255) / 256;

    // ---- GAT layer 1 (C=256) ----
    gemm1_kernel<<<dim3(N / 16, 2), blk, 0, stream>>>(cell_x, W1, h1, N);
    scores_kernel<<<(N + 7) / 8, blk, 0, stream>>>(h1, as1, ad1, sbuf, dbuf, N, 256);
    fill_kernel<<<(N + 255) / 256, blk, 0, stream>>>(mbuf, N, NEG_INF);
    fill_kernel<<<(N + 255) / 256, blk, 0, stream>>>(zbuf, N, 0.f);
    fill_kernel<<<(N * 256 + 255) / 256, blk, 0, stream>>>(o1, N * 256, 0.f);
    edge_max_kernel<<<egrid, blk, 0, stream>>>(esrc, edst, sbuf, dbuf, ew, mbuf, E0, E);
    edge_exp_kernel<<<egrid, blk, 0, stream>>>(esrc, edst, ew, mbuf, zbuf, E0, E);
    aggregate_kernel<<<(E + 7) / 8, blk, 0, stream>>>(esrc, edst, ew, zbuf, h1, o1, E0, E, 256);
    bias_relu_kernel<<<(N * 256 + 255) / 256, blk, 0, stream>>>(o1, b1, N, 256);

    // ---- GAT layer 2 (C=128) ----
    gemm2_wmma_kernel<<<N / 128, blk, 0, stream>>>(o1, W2, h2, N);
    scores_kernel<<<(N + 7) / 8, blk, 0, stream>>>(h2, as2, ad2, sbuf, dbuf, N, 128);
    fill_kernel<<<(N + 255) / 256, blk, 0, stream>>>(mbuf, N, NEG_INF);
    fill_kernel<<<(N + 255) / 256, blk, 0, stream>>>(zbuf, N, 0.f);
    fill_kernel<<<(N * 128 + 255) / 256, blk, 0, stream>>>(o2, N * 128, 0.f);
    edge_max_kernel<<<egrid, blk, 0, stream>>>(esrc, edst, sbuf, dbuf, ew, mbuf, E0, E);
    edge_exp_kernel<<<egrid, blk, 0, stream>>>(esrc, edst, ew, mbuf, zbuf, E0, E);
    aggregate_kernel<<<(E + 7) / 8, blk, 0, stream>>>(esrc, edst, ew, zbuf, h2, o2, E0, E, 128);
    bias_relu_kernel<<<(N * 128 + 255) / 256, blk, 0, stream>>>(o2, b2, N, 128);

    // ---- pool + heads ----
    pool_kernel<<<64, 128, 0, stream>>>(o2, x3);

    head_gemm_kernel<<<(64 * 128 + 255) / 256, blk, 0, stream>>>(x3, cl1_w, cl1_b, t1, 128, 128, 0);
    bn_kernel<<<1, 128, 0, stream>>>(t1, cbn1_g, cbn1_b, 128, 1);
    head_gemm_kernel<<<(64 * 128 + 255) / 256, blk, 0, stream>>>(t1, cl2_w, cl2_b, cemb, 128, 128, 1);

    head_gemm_kernel<<<(64 * 128 + 255) / 256, blk, 0, stream>>>(drug, d1_w, d1_b, u1, 256, 128, 0);
    bn_kernel<<<1, 128, 0, stream>>>(u1, dbn1_g, dbn1_b, 128, 1);
    head_gemm_kernel<<<(64 * 128 + 255) / 256, blk, 0, stream>>>(u1, d2_w, d2_b, demb, 128, 128, 0);
    bn_kernel<<<1, 128, 0, stream>>>(demb, dbn2_g, dbn2_b, 128, 1);

    concat_kernel<<<64, 256, 0, stream>>>(cemb, demb, cat);
    head_gemm_kernel<<<(64 * 128 + 255) / 256, blk, 0, stream>>>(cat, f1_w, f1_b, fm1, 256, 128, 0);
    bn_kernel<<<1, 128, 0, stream>>>(fm1, fbn1_g, fbn1_b, 128, 2);
    head_gemm_kernel<<<(64 * 64 + 255) / 256, blk, 0, stream>>>(fm1, f2_w, f2_b, fm2, 128, 64, 0);
    bn_kernel<<<1, 64, 0, stream>>>(fm2, fbn2_g, fbn2_b, 64, 2);
    final_kernel<<<1, 64, 0, stream>>>(fm2, f3_w, f3_b, (float*)d_out);
}
